// DeepSeekMoE_7438883356685
// MI455X (gfx1250) — matrix operand order repaired and checked by reading, not compiled, not run on hardware
//
#include <hip/hip_runtime.h>

// ---------------------------------------------------------------------------
// MoE (DeepSeek-style): shared expert + top-2 routed experts, bf16 WMMA GEMMs
// Target: gfx1250 (CDNA5), wave32, v_wmma_f32_16x16x32_bf16
// Wave tile 32x32 (4 WMMA/k-step), LDS double buffering; buffer selection via
// pointer ternary so 16B alignment is type-proven -> ds_load/store_b128.
// ---------------------------------------------------------------------------

typedef __attribute__((ext_vector_type(16))) __bf16 v16bf;
typedef __attribute__((ext_vector_type(8)))  float  v8f;

#define T_TOK 4096   // B*S
#define DDIM  1024
#define NEXP  8

#define BM 64        // block tile M (2 waves x 32)
#define BN 128       // block tile N (4 waves x 32)
#define BK 32        // k-slice per stage
#define LDA 40       // halfs per LDS A row (row stride 80B, 16B-aligned)
#define LDB 40
#define ABUF (BM * LDA)   // halfs per A buffer (5120B: 16B-aligned)
#define BBUF (BN * LDB)   // halfs per B buffer (10240B: 16B-aligned)

// WMMA fragment overlay: two 16B-aligned quads -> forces ds_load_b128
union Frag {
    uint4 q[2];
    v16bf v;
};

// round-to-nearest-even f32 -> bf16 (inputs are finite normals)
static __device__ __forceinline__ unsigned short f2bf(float f) {
    unsigned u = __builtin_bit_cast(unsigned, f);
    u += 0x7FFFu + ((u >> 16) & 1u);
    return (unsigned short)(u >> 16);
}

// ---------------------------------------------------------------------------
__global__ void moe_cvt_bf16(const float* __restrict__ src,
                             unsigned short* __restrict__ dst, int n4) {
    int i = blockIdx.x * blockDim.x + threadIdx.x;
    int stride = gridDim.x * blockDim.x;
    const float4* s4 = (const float4*)src;
    for (; i < n4; i += stride) {
        float4 v = s4[i];
        ushort4 o;
        o.x = f2bf(v.x); o.y = f2bf(v.y); o.z = f2bf(v.z); o.w = f2bf(v.w);
        ((ushort4*)dst)[i] = o;
    }
}

// ---------------------------------------------------------------------------
// Router: one wave per token; exact fp32 scores; top-2 + softmax; append
// (token | slot<<31, weight) to per-expert lists.
// ---------------------------------------------------------------------------
__global__ void moe_router(const float* __restrict__ x,
                           const float* __restrict__ Wr,   // [E][D]
                           const float* __restrict__ br,   // [E]
                           int*   __restrict__ counts,     // [E]
                           int*   __restrict__ entries,    // [E][T]
                           float* __restrict__ wlist) {    // [E][T]
    int wave = threadIdx.x >> 5;
    int lane = threadIdx.x & 31;
    int t = blockIdx.x * 8 + wave;
    if (t >= T_TOK) return;

    float acc[NEXP];
#pragma unroll
    for (int e = 0; e < NEXP; ++e) acc[e] = 0.f;

    const float* xp = x + (size_t)t * DDIM;
    for (int d = lane; d < DDIM; d += 32) {
        float xv = xp[d];
#pragma unroll
        for (int e = 0; e < NEXP; ++e) acc[e] += xv * Wr[e * DDIM + d];
    }
#pragma unroll
    for (int e = 0; e < NEXP; ++e) {
#pragma unroll
        for (int off = 16; off > 0; off >>= 1)
            acc[e] += __shfl_xor(acc[e], off, 32);
    }
    if (lane == 0) {
        float s[NEXP];
#pragma unroll
        for (int e = 0; e < NEXP; ++e) s[e] = acc[e] + br[e];
        int i0 = 0;
#pragma unroll
        for (int e = 1; e < NEXP; ++e) if (s[e] > s[i0]) i0 = e;   // ties: lowest idx
        int i1 = -1;
#pragma unroll
        for (int e = 0; e < NEXP; ++e)
            if (e != i0 && (i1 < 0 || s[e] > s[i1])) i1 = e;
        float z  = expf(s[i1] - s[i0]);     // s[i0] is the max
        float w0 = 1.f / (1.f + z);
        float w1 = z * w0;
        int p0 = atomicAdd(&counts[i0], 1);
        entries[i0 * T_TOK + p0] = t;                         // slot 0
        wlist  [i0 * T_TOK + p0] = w0;
        int p1 = atomicAdd(&counts[i1], 1);
        entries[i1 * T_TOK + p1] = t | (int)0x80000000u;      // slot 1
        wlist  [i1 * T_TOK + p1] = w1;
    }
}

// ---------------------------------------------------------------------------
// Tiled WMMA GEMM: C[m,n] = sum_k A[m,k] * W[n,k]  (+ bias, + gate scaling)
// 256 threads = 8 waves in 2(M) x 4(N) grid; wave tile 32x32 (4 WMMA/k-step);
// double-buffered LDS + register prefetch -> 1 barrier per k-step.
// EXPERT=false : A rows = tokens; write acc+bias to out[t*D+n]
// EXPERT=true  : A rows gathered via expert list; write w*(acc+bias) to
//                routed[slot][tok][n] (each (tok,slot) written exactly once)
// ---------------------------------------------------------------------------
template <bool EXPERT>
__global__ void moe_gemm(const unsigned short* __restrict__ Amat, // bf16 [T][D]
                         const unsigned short* __restrict__ Wmat, // bf16 [(E)][D][D]
                         const float* __restrict__ bias,          // [(E)][D]
                         float* __restrict__ outbuf,              // out or routed[2][T][D]
                         const int* __restrict__ counts,
                         const int* __restrict__ entries,
                         const float* __restrict__ wlist) {
    const int e     = EXPERT ? blockIdx.z : 0;
    const int mBase = blockIdx.y * BM;
    const int nBase = blockIdx.x * BN;
    const int cnt   = EXPERT ? counts[e] : T_TOK;
    if (mBase >= cnt) return;

    const unsigned short* W = Wmat + (size_t)e * DDIM * DDIM;
    const float* bvec = bias + (size_t)e * DDIM;

    __shared__ __align__(16) unsigned short As[2 * ABUF];
    __shared__ __align__(16) unsigned short Bs[2 * BBUF];
    __shared__ int toks[BM];

    const int tid  = threadIdx.x;
    const int lane = tid & 31;
    const int wave = tid >> 5;
    const int wm   = wave >> 2;   // 0..1
    const int wn   = wave & 3;    // 0..3
    const int half = lane >> 4;   // 0|1
    const int l16  = lane & 15;

    if (EXPERT) {
        if (tid < BM) {
            int p  = mBase + tid;
            int en = (p < cnt) ? entries[e * T_TOK + p] : 0;
            toks[tid] = en & 0x7FFFFFFF;
        }
        __syncthreads();
    }

    // global->LDS load mappings (uint4 = 8 halfs)
    const int lrA = tid >> 2;            // 0..63
    const int kcA = (tid & 3) * 8;       // 0,8,16,24
    const int iB0 = tid,       lrB0 = iB0 >> 2, kcB0 = (iB0 & 3) * 8;  // rows 0..63
    const int iB1 = tid + 256, lrB1 = iB1 >> 2, kcB1 = (iB1 & 3) * 8;  // rows 64..127

    const int growA = EXPERT ? toks[lrA] : (mBase + lrA);
    const unsigned short* aptr  = Amat + (size_t)growA * DDIM + kcA;
    const unsigned short* bptr0 = W + (size_t)(nBase + lrB0) * DDIM + kcB0;
    const unsigned short* bptr1 = W + (size_t)(nBase + lrB1) * DDIM + kcB1;

    // LDS store pointers, one per buffer (16B alignment carried by uint4*)
    uint4* const sA0  = (uint4*)&As[lrA  * LDA + kcA ];
    uint4* const sA1  = (uint4*)&As[ABUF + lrA  * LDA + kcA ];
    uint4* const sB00 = (uint4*)&Bs[lrB0 * LDB + kcB0];
    uint4* const sB01 = (uint4*)&Bs[BBUF + lrB0 * LDB + kcB0];
    uint4* const sB10 = (uint4*)&Bs[lrB1 * LDB + kcB1];
    uint4* const sB11 = (uint4*)&Bs[BBUF + lrB1 * LDB + kcB1];

    // fragment read pointers, one per buffer (dword units)
    // A row m at halfs m*LDA (80B, 16B-aligned); dword groups at +ka, +8+ka
    const int frA0 = ((wm * 32 + l16) * LDA) >> 1;   // dwords; +16 rows for a1
    const int frB0 = ((wn * 32 + l16) * LDB) >> 1;   // dwords; +16 rows for b1
    const int ka   = half * 4;   // dwords
    const int kb   = half * 4;   // dwords
    const unsigned* const rA0 = (const unsigned*)As + frA0;
    const unsigned* const rA1 = (const unsigned*)As + (ABUF >> 1) + frA0;
    const unsigned* const rB0 = (const unsigned*)Bs + frB0;
    const unsigned* const rB1 = (const unsigned*)Bs + (BBUF >> 1) + frB0;

    // stage k=0 into buffer 0
    uint4 pa  = *(const uint4*)(aptr);
    uint4 pb0 = *(const uint4*)(bptr0);
    uint4 pb1 = *(const uint4*)(bptr1);
    *sA0  = pa;
    *sB00 = pb0;
    *sB10 = pb1;

    v8f c00 = (v8f){}, c01 = (v8f){}, c10 = (v8f){}, c11 = (v8f){};

    const int NK = DDIM / BK;      // 32

#pragma clang loop unroll(disable)
    for (int kk = 0; kk < NK; ++kk) {
        const int  nb   = kk & 1;       // 0: reading buf0, next is buf1
        const bool more = (kk + 1) < NK;
        if (more) {  // register prefetch of next k-slice (waits overlap WMMAs)
            int k0 = (kk + 1) * BK;
            pa  = *(const uint4*)(aptr  + k0);
            pb0 = *(const uint4*)(bptr0 + k0);
            pb1 = *(const uint4*)(bptr1 + k0);
        }
        __syncthreads();  // current buffer visible; prior reads of other done

        // --- fragments (ISA 7.12.2 16-bit layouts), 16B quads -> ds_load_b128
        // A 16x32: lane<16 holds K {0..7,16..23}; lane>=16 holds {8..15,24..31}
        // B 32x16: lane<16 holds K {0..7,16..23} of its column; lane>=16 {8..15,24..31}
        const unsigned* pA = nb ? rA1 : rA0;
        const unsigned* pB = nb ? rB1 : rB0;
        Frag a0, a1, b0, b1;
        a0.q[0] = *(const uint4*)(pA + ka);
        a0.q[1] = *(const uint4*)(pA + 8 + ka);
        b0.q[0] = *(const uint4*)(pB + kb);
        b0.q[1] = *(const uint4*)(pB + kb + 8);
        b1.q[0] = *(const uint4*)(pB + ((16 * LDB) >> 1) + kb);
        b1.q[1] = *(const uint4*)(pB + ((16 * LDB) >> 1) + kb + 8);
        a1.q[0] = *(const uint4*)(pA + ((16 * LDA) >> 1) + ka);
        a1.q[1] = *(const uint4*)(pA + ((16 * LDA) >> 1) + 8 + ka);

        c00 = __builtin_amdgcn_wmma_f32_16x16x32_bf16(
            false, a0.v, false, b0.v, (short)0, c00, false, false);
        c01 = __builtin_amdgcn_wmma_f32_16x16x32_bf16(
            false, a0.v, false, b1.v, (short)0, c01, false, false);
        c10 = __builtin_amdgcn_wmma_f32_16x16x32_bf16(
            false, a1.v, false, b0.v, (short)0, c10, false, false);
        c11 = __builtin_amdgcn_wmma_f32_16x16x32_bf16(
            false, a1.v, false, b1.v, (short)0, c11, false, false);

        if (more) {  // fill the other buffer; readers sync at next barrier
            *(nb ? sA0  : sA1 ) = pa;
            *(nb ? sB00 : sB01) = pb0;
            *(nb ? sB10 : sB11) = pb1;
        }
    }

    // --- epilogue: frag (fi,fj): m = wm*32+fi*16+half*8+r, n = wn*32+fj*16+l16
    const int gn0 = nBase + wn * 32 + l16;
    const int gn1 = gn0 + 16;
    const float bn0 = bvec[gn0];
    const float bn1 = bvec[gn1];
    const v8f cr0[2] = {c00, c01};   // rows fi=0
    const v8f cr1[2] = {c10, c11};   // rows fi=1
#pragma unroll
    for (int fi = 0; fi < 2; ++fi) {
        const v8f* cr = fi ? cr1 : cr0;
#pragma unroll
        for (int r = 0; r < 8; ++r) {
            int p = mBase + wm * 32 + fi * 16 + half * 8 + r;
            if (EXPERT) {
                if (p < cnt) {
                    int en   = entries[e * T_TOK + p];
                    int tok  = en & 0x7FFFFFFF;
                    int slot = ((unsigned)en) >> 31;
                    float w  = wlist[e * T_TOK + p];
                    float* dst = outbuf + (size_t)slot * T_TOK * DDIM +
                                 (size_t)tok * DDIM;
                    dst[gn0] = w * (cr[0][r] + bn0);
                    dst[gn1] = w * (cr[1][r] + bn1);
                }
            } else {
                float* dst = outbuf + (size_t)p * DDIM;
                dst[gn0] = cr[0][r] + bn0;
                dst[gn1] = cr[1][r] + bn1;
            }
        }
    }
}

// ---------------------------------------------------------------------------
// out += routed_slot0 + routed_slot1   (fixed order -> deterministic)
// ---------------------------------------------------------------------------
__global__ void moe_combine(float* __restrict__ out,
                            const float* __restrict__ routed, int n4) {
    int i = blockIdx.x * blockDim.x + threadIdx.x;
    int stride = gridDim.x * blockDim.x;
    const float4* r0 = (const float4*)routed;
    const float4* r1 = (const float4*)(routed + (size_t)n4 * 4);
    float4* o = (float4*)out;
    for (; i < n4; i += stride) {
        float4 a = o[i], u = r0[i], v = r1[i];
        a.x += u.x + v.x; a.y += u.y + v.y;
        a.z += u.z + v.z; a.w += u.w + v.w;
        o[i] = a;
    }
}

// ---------------------------------------------------------------------------
extern "C" void kernel_launch(void* const* d_in, const int* in_sizes, int n_in,
                              void* d_out, int out_size, void* d_ws, size_t ws_size,
                              hipStream_t stream) {
    (void)in_sizes; (void)n_in; (void)out_size;
    const float* x    = (const float*)d_in[0];   // [T][D]
    const float* Ws   = (const float*)d_in[1];   // [D][D]
    const float* bs   = (const float*)d_in[2];   // [D]
    const float* We   = (const float*)d_in[3];   // [E][D][D]
    const float* be   = (const float*)d_in[4];   // [E][D]
    const float* Wr   = (const float*)d_in[5];   // [E][D]
    const float* br   = (const float*)d_in[6];   // [E]
    float* out = (float*)d_out;

    const size_t T = T_TOK, D = DDIM, E = NEXP;

    // workspace carve-out (256B aligned)
    char* ws = (char*)d_ws;
    size_t off = 0;
    auto carve = [&](size_t bytes) {
        size_t o = off;
        off = (off + bytes + 255) & ~(size_t)255;
        return o;
    };
    unsigned short* xb   = (unsigned short*)(ws + carve(T * D * 2));
    unsigned short* wsb  = (unsigned short*)(ws + carve(D * D * 2));
    unsigned short* web  = (unsigned short*)(ws + carve(E * D * D * 2));
    float*          rout = (float*)         (ws + carve(2 * T * D * 4));
    int*            cnts = (int*)           (ws + carve(E * 4));
    int*            ents = (int*)           (ws + carve(E * T * 4));
    float*          wlst = (float*)         (ws + carve(E * T * 4));
    if (off > ws_size) return;  // workspace too small: nothing safe to do

    hipMemsetAsync(cnts, 0, E * sizeof(int), stream);

    // 1) bf16 conversions
    moe_cvt_bf16<<<4096, 256, 0, stream>>>(x,  xb,  (int)(T * D / 4));
    moe_cvt_bf16<<<1024, 256, 0, stream>>>(Ws, wsb, (int)(D * D / 4));
    moe_cvt_bf16<<<8192, 256, 0, stream>>>(We, web, (int)(E * D * D / 4));

    // 2) routing (exact fp32 selection)
    moe_router<<<T_TOK / 8, 256, 0, stream>>>(x, Wr, br, cnts, ents, wlst);

    // 3) shared-expert GEMM -> d_out
    moe_gemm<false><<<dim3(DDIM / BN, T_TOK / BM), 256, 0, stream>>>(
        xb, wsb, bs, out, cnts, ents, wlst);

    // 4) routed-expert GEMMs (gathered rows; empty tiles exit immediately)
    moe_gemm<true><<<dim3(DDIM / BN, T_TOK / BM, NEXP), 256, 0, stream>>>(
        xb, web, be, rout, cnts, ents, wlst);

    // 5) combine
    moe_combine<<<4096, 256, 0, stream>>>(out, rout, (int)(T * D / 4));
}
